// SelfTransFormer_86526411145610
// MI455X (gfx1250) — matrix operand order repaired and checked by reading, not compiled, not run on hardware
//
#include <hip/hip_runtime.h>
#include <hip/hip_bf16.h>

typedef __attribute__((ext_vector_type(2)))  float        v2f;
typedef __attribute__((ext_vector_type(8)))  float        v8f;
typedef __attribute__((ext_vector_type(8)))  unsigned int v8u;
typedef __attribute__((ext_vector_type(16))) __bf16       v16bf;
typedef __attribute__((ext_vector_type(4)))  unsigned int v4u;
typedef __attribute__((ext_vector_type(8)))  int          v8i;
typedef __attribute__((ext_vector_type(4)))  int          v4i;

#define G_     8
#define NG_    32
#define HEADS  8
#define BATCH  8
#define CIN    512
#define COUT   1024
#define PP     2048
#define DH     128
#define EPS_   1e-5f
#define SLOPE_ 0.1f

#define LDT    132                 // padded LDS row pitch (ushorts) for 128-elem rows
#define TILE_USH (128 * LDT)       // one 128x128 bf16 tile in LDS (padded)

__device__ __forceinline__ unsigned short f2bf(float f) {
  unsigned int u = __builtin_bit_cast(unsigned int, f);
  u += 0x7FFFu + ((u >> 16) & 1u);
  return (unsigned short)(u >> 16);
}

// 16-bit A-matrix (16xK) k index held by VGPR r for a lane half (contiguous: k = 16*half + 2*r)
__device__ __forceinline__ int kmapA(int r, int half) {
  return 16 * half + 2 * r;
}

// ---------------------------------------------------------------------------
// TDM: 2D bf16 tile load (128 rows x 128 elems) global->LDS with row padding.
// pad_interval=5 (64 DWORDs = one 256B row), pad_amount=1 (2 DWORDs = 8B)
// reproduces the 132-ushort padded LDS pitch. Tracked by TENSORcnt.
// D# packing per CDNA5 ISA.
// This toolchain exposes the 6-arg builtin:
//   (uint32x4 g0, int32x8 g1, int32x4 g2, int32x4 g3, int32x8, i32 cpol)
// ---------------------------------------------------------------------------
__device__ __forceinline__ void tdm_load_tile_2d(unsigned lds_addr,
                                                 const void* gptr,
                                                 unsigned stride_elems) {
  const unsigned long long ga = (unsigned long long)(uintptr_t)gptr;
  v4u g0;
  g0[0] = 1u;                                            // count=1, user mode
  g0[1] = lds_addr;                                      // lds_addr [63:32]
  g0[2] = (unsigned)(ga & 0xFFFFFFFFu);                  // global_addr lo
  g0[3] = (unsigned)((ga >> 32) & 0x01FFFFFFu) | (2u << 30); // addr hi | type=2
  v8i g1;
  g1[0] = (int)((1u << 16) | (1u << 20) | (5u << 22) | (1u << 25));
  //        data_size=2B | pad_enable | pad_interval=64dw | pad_amount=2dw
  g1[1] = (int)(128u << 16);   // tensor_dim0 = 128 (bits [79:48], low half)
  g1[2] = (int)(128u << 16);   // tensor_dim1 = 128 (bits [111:80], low half)
  g1[3] = (int)(128u << 16);   // tile_dim0   = 128 (bits [127:112])
  g1[4] = 128;                 // tile_dim1   = 128 (bits [143:128])
  g1[5] = (int)stride_elems;   // tensor_dim0_stride low 32 (bits [207:160])
  g1[6] = 0;
  g1[7] = 0;
  const v4i z4 = {0, 0, 0, 0};                 // groups 2/3 unused (2D tensor)
  const v8i z8 = {0, 0, 0, 0, 0, 0, 0, 0};
  __builtin_amdgcn_tensor_load_to_lds(g0, g1, z4, z4, z8, 0);
}

// ---------------------------------------------------------------------------
// Kernel 1: grouped 1x1 conv + bias + GroupNorm + LeakyReLU, fp32 WMMA.
// grid.x = BATCH*NG (one WG per (b, norm-group of 32 channels)), grid.y = branch
// Writes bf16: branch 0 -> Q as [B, C, P]; branch 1/2 -> K/V as [B, H, P, d].
// ---------------------------------------------------------------------------
__global__ __launch_bounds__(256) void qkv_branch_kernel(
    const float* __restrict__ x,
    const float* __restrict__ wq, const float* __restrict__ bq,
    const float* __restrict__ gw1, const float* __restrict__ gb1,
    const float* __restrict__ wk, const float* __restrict__ bk,
    const float* __restrict__ gw2, const float* __restrict__ gb2,
    const float* __restrict__ wv, const float* __restrict__ bv,
    const float* __restrict__ gw3, const float* __restrict__ gb3,
    unsigned short* __restrict__ Qb, unsigned short* __restrict__ Kb,
    unsigned short* __restrict__ Vb)
{
  const int tid  = threadIdx.x;
  const int wid  = tid >> 5;
  const int lane = tid & 31;
  const int ln   = lane & 15;
  const int half = lane >> 4;

  const int branch = blockIdx.y;
  const float *w, *bias, *gw, *gb;
  unsigned short* outp;
  if (branch == 0)      { w = wq; bias = bq; gw = gw1; gb = gb1; outp = Qb; }
  else if (branch == 1) { w = wk; bias = bk; gw = gw2; gb = gb2; outp = Kb; }
  else                  { w = wv; bias = bv; gw = gw3; gb = gb3; outp = Vb; }

  const int b  = blockIdx.x / NG_;
  const int ng = blockIdx.x % NG_;
  const int g  = ng >> 2;          // conv group (4 norm-groups per conv group)
  const int o0 = ng * 32;          // global output-channel base

  extern __shared__ __align__(16) char smem[];
  float* yl     = (float*)smem;            // [32][2049] padded tile, 256KB+
  float* Wl     = yl + 32 * 2049;          // [32][64]
  float* bias_l = Wl + 32 * 64;
  float* gwl    = bias_l + 32;
  float* gbl    = gwl + 32;
  float* red    = gbl + 32;                // [512]

  // stage weights / affine params
  for (int t = tid; t < 32 * 64; t += 256) Wl[t] = w[o0 * 64 + t];
  if (tid < 32) {
    bias_l[tid] = bias[o0 + tid];
    gwl[tid]    = gw[o0 + tid];
    gbl[tid]    = gb[o0 + tid];
  }
  __syncthreads();

  const float* xg = x + ((size_t)b * CIN + g * 64) * PP;

  // Hoist A (weight) fragments: 2 M-tiles x 16 k-steps, f32 16x16x4 layout
  v2f afr[2][16];
#pragma unroll
  for (int mt = 0; mt < 2; ++mt)
#pragma unroll
    for (int kt = 0; kt < 16; ++kt) {
      const int k = kt * 4 + 2 * half;           // K = r + 2*half, r in {0,1}
      afr[mt][kt] = *(const v2f*)&Wl[(mt * 16 + ln) * 64 + k];
    }

  float sum = 0.f, sumsq = 0.f;

  // each wave owns a 256-column slab: 16 N-tiles x (2 M-tiles)
  for (int nt = 0; nt < 16; ++nt) {
    const int p = wid * 256 + nt * 16 + ln;
    v8f acc0 = {}, acc1 = {};
#pragma unroll
    for (int kt = 0; kt < 16; ++kt) {
      const int k = kt * 4 + 2 * half;
      v2f bf;
      bf.x = xg[(size_t)k * PP + p];
      bf.y = xg[(size_t)(k + 1) * PP + p];
      acc0 = __builtin_amdgcn_wmma_f32_16x16x4_f32(false, afr[0][kt],
                                                   false, bf, (short)0, acc0,
                                                   false, false);
      acc1 = __builtin_amdgcn_wmma_f32_16x16x4_f32(false, afr[1][kt],
                                                   false, bf, (short)0, acc1,
                                                   false, false);
    }
#pragma unroll
    for (int mt = 0; mt < 2; ++mt) {
      v8f a = (mt == 0) ? acc0 : acc1;
#pragma unroll
      for (int r = 0; r < 8; ++r) {
        const int row = mt * 16 + r + 8 * half;     // D: M = r + 8*half
        const float v = a[r] + bias_l[row];
        yl[row * 2049 + p] = v;
        sum += v;
        sumsq += v * v;
      }
    }
  }

  // group statistics over 32*2048 elements
  red[tid] = sum;
  red[256 + tid] = sumsq;
  __syncthreads();
  for (int s = 128; s > 0; s >>= 1) {
    if (tid < s) { red[tid] += red[tid + s]; red[256 + tid] += red[256 + tid + s]; }
    __syncthreads();
  }
  const float mean = red[0] * (1.0f / 65536.0f);
  const float var  = red[256] * (1.0f / 65536.0f) - mean * mean;
  const float rstd = rsqrtf(var + EPS_);

  // normalize + LeakyReLU + bf16 store (layout per branch)
  if (branch == 0) {
    for (int e = tid; e < 32 * 2048; e += 256) {
      const int row = e >> 11, p = e & 2047;
      float v = (yl[row * 2049 + p] - mean) * rstd * gwl[row] + gbl[row];
      v = (v >= 0.f) ? v : SLOPE_ * v;
      outp[(((size_t)b * COUT + o0 + row) << 11) + p] = f2bf(v);
    }
  } else {
    const int h  = o0 >> 7;
    const int c0 = o0 & 127;
    for (int e = tid; e < 32 * 2048; e += 256) {
      const int cc = e & 31, p = e >> 5;
      float v = (yl[cc * 2049 + p] - mean) * rstd * gwl[cc] + gbl[cc];
      v = (v >= 0.f) ? v : SLOPE_ * v;
      outp[((((size_t)b * HEADS + h) << 11) + p) * DH + c0 + cc] = f2bf(v);
    }
  }
}

// ---------------------------------------------------------------------------
// Kernel 2: flash attention. scores = scale * K^T V, softmax over j,
// O[i,c] = sum_j P[i,j] Q^T[j,c]. bf16 WMMA 16x16x32, f32 accumulate.
// TDM double-buffered tile staging (V and Q tiles DMA'd by wave 0 while all
// waves run WMMA on the other buffer; TENSORcnt + barrier publish).
// grid.x = P/128 (i tiles), grid.y = B*H. 8 waves, 16 i-rows per wave.
// ---------------------------------------------------------------------------
__global__ __launch_bounds__(256) void flash_attn_kernel(
    const unsigned short* __restrict__ Qbf,
    const unsigned short* __restrict__ Kbf,
    const unsigned short* __restrict__ Vbf,
    float* __restrict__ out)
{
  const int tid  = threadIdx.x;
  const int wid  = tid >> 5;
  const int lane = tid & 31;
  const int ln   = lane & 15;
  const int half = lane >> 4;

  const int ibase = blockIdx.x * 128;
  const int bh = blockIdx.y, b = bh >> 3, h = bh & 7;

  const unsigned short* Kt = Kbf + (size_t)bh * PP * DH;   // [P][d]
  const unsigned short* Vt = Vbf + (size_t)bh * PP * DH;   // [P][d]
  const unsigned short* Qc = Qbf + ((size_t)b * COUT + h * DH) * PP; // [d][P]

  extern __shared__ __align__(16) char smem[];
  unsigned short* Vl0 = (unsigned short*)smem;     // 2 x [128][132] V tiles
  unsigned short* Ql0 = Vl0 + 2 * TILE_USH;        // 2 x [128][132] Q tiles
  unsigned short* Pl  = Ql0 + 2 * TILE_USH;        // 8 x [16][132] P strips
  unsigned short* Pw  = Pl + wid * 16 * LDT;

  // Hoist K^T A-fragments (depends only on i): 4 k-steps of 32 channels
  v8u aK[4];
  {
    const unsigned short* kr = Kt + (size_t)(ibase + wid * 16 + ln) * DH;
#pragma unroll
    for (int kt = 0; kt < 4; ++kt)
#pragma unroll
      for (int r = 0; r < 8; ++r)
        aK[kt][r] = *(const unsigned int*)(kr + kt * 32 + kmapA(r, half));
  }

  float m[8], l[8];
  v8f O[8];
#pragma unroll
  for (int r = 0; r < 8; ++r) { m[r] = -1e30f; l[r] = 0.f; }
#pragma unroll
  for (int nt = 0; nt < 8; ++nt) O[nt] = {};

  const float scale = 0.08838834764831845f;  // 1/sqrt(128)
  const int NT = PP / 128;

  // prologue: DMA first tiles into buffer 0
  if (wid == 0) {
    tdm_load_tile_2d((unsigned)(uintptr_t)Vl0, Vt, DH);
    tdm_load_tile_2d((unsigned)(uintptr_t)Ql0, Qc, PP);
  }

  for (int jt = 0; jt < NT; ++jt) {
    const int buf = jt & 1;
    if (wid == 0) __builtin_amdgcn_s_wait_tensorcnt(0);
    __syncthreads();   // publish DMA'd tiles; retire reads of the other buffer

    if (wid == 0 && jt + 1 < NT) {   // prefetch next tiles into other buffer
      const int jn = (jt + 1) * 128;
      tdm_load_tile_2d((unsigned)(uintptr_t)(Vl0 + (buf ^ 1) * TILE_USH),
                       Vt + (size_t)jn * DH, DH);
      tdm_load_tile_2d((unsigned)(uintptr_t)(Ql0 + (buf ^ 1) * TILE_USH),
                       Qc + jn, PP);
    }

    const unsigned short* Vl = Vl0 + buf * TILE_USH;
    const unsigned short* Ql = Ql0 + buf * TILE_USH;

    // S = scale * K^T V : per wave 16 x 128 (8 N-tiles)
    v8f s[8];
#pragma unroll
    for (int nt = 0; nt < 8; ++nt) {
      v8f acc = {};
      const unsigned short* vrow = Vl + (nt * 16 + ln) * LDT;
#pragma unroll
      for (int kt = 0; kt < 4; ++kt) {
        v8u bv;
#pragma unroll
        for (int r = 0; r < 8; ++r)
          bv[r] = *(const unsigned int*)(vrow + kt * 32 + 16 * half + 2 * r);
        acc = __builtin_amdgcn_wmma_f32_16x16x32_bf16(
            false, __builtin_bit_cast(v16bf, aK[kt]),
            false, __builtin_bit_cast(v16bf, bv),
            (short)0, acc, false, false);
      }
#pragma unroll
      for (int e = 0; e < 8; ++e) acc[e] *= scale;
      s[nt] = acc;
    }

    // online softmax; row r+8*half owned by this lane's 16-group
#pragma unroll
    for (int r = 0; r < 8; ++r) {
      float v = s[0][r];
#pragma unroll
      for (int nt = 1; nt < 8; ++nt) v = fmaxf(v, s[nt][r]);
      for (int off = 1; off < 16; off <<= 1)
        v = fmaxf(v, __shfl_xor(v, off, 32));
      const float mn = fmaxf(m[r], v);
      const float alpha = __expf(m[r] - mn);
      m[r] = mn;
      float rs = 0.f;
#pragma unroll
      for (int nt = 0; nt < 8; ++nt) {
        const float p = __expf(s[nt][r] - mn);
        s[nt][r] = p;
        rs += p;
      }
      for (int off = 1; off < 16; off <<= 1) rs += __shfl_xor(rs, off, 32);
      l[r] = l[r] * alpha + rs;
#pragma unroll
      for (int nt = 0; nt < 8; ++nt) O[nt][r] *= alpha;
    }

    // P -> bf16 via wave-private LDS strip (transpose to A-frag layout)
#pragma unroll
    for (int nt = 0; nt < 8; ++nt)
#pragma unroll
      for (int r = 0; r < 8; ++r)
        Pw[(r + 8 * half) * LDT + nt * 16 + ln] = f2bf(s[nt][r]);
    asm volatile("s_wait_dscnt 0x0" ::: "memory");  // wave-local DS RAW

    v8u aP[4];
#pragma unroll
    for (int kt = 0; kt < 4; ++kt)
#pragma unroll
      for (int r = 0; r < 8; ++r)
        aP[kt][r] = *(const unsigned int*)(Pw + ln * LDT + kt * 32 + kmapA(r, half));

    // O += P * Q^T
#pragma unroll
    for (int nt = 0; nt < 8; ++nt) {
      const unsigned short* qrow = Ql + (nt * 16 + ln) * LDT;
      v8f acc = O[nt];
#pragma unroll
      for (int kt = 0; kt < 4; ++kt) {
        v8u bq;
#pragma unroll
        for (int r = 0; r < 8; ++r)
          bq[r] = *(const unsigned int*)(qrow + kt * 32 + 16 * half + 2 * r);
        acc = __builtin_amdgcn_wmma_f32_16x16x32_bf16(
            false, __builtin_bit_cast(v16bf, aP[kt]),
            false, __builtin_bit_cast(v16bf, bq),
            (short)0, acc, false, false);
      }
      O[nt] = acc;
    }
  }

  // epilogue: divide by row sums; D-frag rows are consecutive i in memory
  float linv[8];
#pragma unroll
  for (int r = 0; r < 8; ++r) linv[r] = 1.0f / l[r];
  const int i0 = ibase + wid * 16 + 8 * half;
#pragma unroll
  for (int nt = 0; nt < 8; ++nt) {
    v8f o = O[nt];
#pragma unroll
    for (int r = 0; r < 8; ++r) o[r] *= linv[r];
    const int c = h * DH + nt * 16 + ln;
    *(v8f*)&out[((size_t)b * COUT + c) * PP + i0] = o;
  }
}

// ---------------------------------------------------------------------------
extern "C" void kernel_launch(void* const* d_in, const int* in_sizes, int n_in,
                              void* d_out, int out_size, void* d_ws, size_t ws_size,
                              hipStream_t stream) {
  const float* x   = (const float*)d_in[0];
  const float* wq  = (const float*)d_in[1];
  const float* bq  = (const float*)d_in[2];
  const float* gw1 = (const float*)d_in[3];
  const float* gb1 = (const float*)d_in[4];
  const float* wk  = (const float*)d_in[5];
  const float* bk  = (const float*)d_in[6];
  const float* gw2 = (const float*)d_in[7];
  const float* gb2 = (const float*)d_in[8];
  const float* wv  = (const float*)d_in[9];
  const float* bv  = (const float*)d_in[10];
  const float* gw3 = (const float*)d_in[11];
  const float* gb3 = (const float*)d_in[12];
  float* out = (float*)d_out;

  const size_t nE = (size_t)BATCH * COUT * PP;
  unsigned short* Qb = (unsigned short*)d_ws;
  unsigned short* Kb = Qb + nE;
  unsigned short* Vb = Kb + nE;

  const size_t smem1 = (size_t)(32 * 2049 + 32 * 64 + 96 + 512) * sizeof(float);
  qkv_branch_kernel<<<dim3(BATCH * NG_, 3), 256, smem1, stream>>>(
      x, wq, bq, gw1, gb1, wk, bk, gw2, gb2, wv, bv, gw3, gb3, Qb, Kb, Vb);

  const size_t smem2 = (size_t)(4 * TILE_USH + 8 * 16 * LDT) * sizeof(unsigned short);
  flash_attn_kernel<<<dim3(PP / 128, BATCH * HEADS), 256, smem2, stream>>>(
      Qb, Kb, Vb, out);
}
